// TGCN_MessageCoupling_76012331204906
// MI455X (gfx1250) — compile-verified, matrix operand order checked by reading, not run on hardware
//
#include <hip/hip_runtime.h>
#include <hip/hip_bf16.h>

typedef __attribute__((ext_vector_type(16))) __bf16 v16bf;
typedef __attribute__((ext_vector_type(4)))  __bf16 v4bf;
typedef __attribute__((ext_vector_type(8)))  float  v8f;

__device__ __forceinline__ v8f wmma_bf16(v16bf a, v16bf b, v8f c) {
  return __builtin_amdgcn_wmma_f32_16x16x32_bf16(false, a, false, b, (short)0, c, false, false);
}

__device__ __forceinline__ void atomAddF(float* p, float v) { unsafeAtomicAdd(p, v); }

// B fragment for row-major bf16 weight [O x strideK] (strideK multiple of 32, 16B aligned):
// lane column c, K-step ks. 16x32 bf16 fragment = two contiguous 8-element runs.
__device__ __forceinline__ v16bf loadB(const __bf16* __restrict__ W, int strideK,
                                       int c, int ks, int half) {
  const __bf16* p = W + (long)c * strideK + ks * 32 + half * 8;
  v16bf B;
  ((uint4*)&B)[0] = *(const uint4*)p;
  ((uint4*)&B)[1] = *(const uint4*)(p + 16);
  return B;
}

// ---------------- utility kernels ----------------
__global__ void k_fill(float* __restrict__ p, float v, long n) {
  long i = (long)blockIdx.x * blockDim.x + threadIdx.x;
  if (i < n) p[i] = v;
}
__global__ void k_filli(int* __restrict__ p, int v, int n) {
  int i = blockIdx.x * blockDim.x + threadIdx.x;
  if (i < n) p[i] = v;
}

// integer in-degree count (excludes self-loops)
__global__ void k_cnt(const int* __restrict__ col, int* __restrict__ cnt, int E) {
  for (int e = blockIdx.x * blockDim.x + threadIdx.x; e < E; e += gridDim.x * blockDim.x)
    atomicAdd(&cnt[col[e]], 1);
}
// dinv = (cnt + 1)^-1/2   (+1 = self loop)
__global__ void k_dinv(const int* __restrict__ cnt, float* __restrict__ dinv, int n) {
  int i = blockIdx.x * blockDim.x + threadIdx.x;
  if (i < n) dinv[i] = rsqrtf((float)(cnt[i] + 1));
}

// ---------------- block-scan (exclusive) for CSR offsets ----------------
#define SCAN_B 1024
__global__ void k_scan1(const int* __restrict__ cnt, int* __restrict__ loc,
                        int* __restrict__ bsum, int n) {
  __shared__ int sh[SCAN_B];
  int t = threadIdx.x, i = blockIdx.x * SCAN_B + t;
  int v = (i < n) ? cnt[i] : 0;
  sh[t] = v; __syncthreads();
  for (int off = 1; off < SCAN_B; off <<= 1) {
    int add = (t >= off) ? sh[t - off] : 0;
    __syncthreads();
    sh[t] += add;
    __syncthreads();
  }
  if (i < n) loc[i] = sh[t] - v;
  if (t == SCAN_B - 1) bsum[blockIdx.x] = sh[t];
}
__global__ void k_scan2(int* __restrict__ bsum, int nb) {
  __shared__ int sh[SCAN_B];
  int t = threadIdx.x;
  int v = (t < nb) ? bsum[t] : 0;
  sh[t] = v; __syncthreads();
  for (int off = 1; off < SCAN_B; off <<= 1) {
    int add = (t >= off) ? sh[t - off] : 0;
    __syncthreads();
    sh[t] += add;
    __syncthreads();
  }
  if (t < nb) bsum[t] = sh[t] - v;
}
__global__ void k_scan3(const int* __restrict__ loc, const int* __restrict__ bsum,
                        const int* __restrict__ cnt, int* __restrict__ offs, int n) {
  int i = blockIdx.x * blockDim.x + threadIdx.x;
  if (i < n) {
    int o = loc[i] + bsum[i / SCAN_B];
    offs[i] = o;
    if (i == n - 1) offs[n] = o + cnt[i];
  }
}

// CSR fill: edges grouped by destination; per-edge src + premultiplied norm
__global__ void k_csrfill(const int* __restrict__ row, const int* __restrict__ col,
                          const float* __restrict__ dinv, const int* __restrict__ offs,
                          int* __restrict__ cur, int* __restrict__ srcs,
                          float* __restrict__ nrm, int E) {
  for (int e = blockIdx.x * blockDim.x + threadIdx.x; e < E; e += gridDim.x * blockDim.x) {
    int s = row[e], d = col[e];
    int p = offs[d] + atomicAdd(&cur[d], 1);
    srcs[p] = s;
    nrm[p] = dinv[s] * dinv[d];
  }
}

// weight copies, bf16, K padded to multiples of 32 for b128 fragment loads
__global__ void k_prep(const float* g, const float* ih, const float* hh,
                       const float* e1, const float* p1,
                       __bf16* og, __bf16* oih, __bf16* ohh, __bf16* oe1, __bf16* op1) {
  int i = blockIdx.x * blockDim.x + threadIdx.x;
  if (i < 64 * 32)   { int o = i >> 5, k = i & 31; og[i]  = (k < 16)  ? (__bf16)g[o * 16 + k]  : (__bf16)0.0f; }
  if (i < 192 * 64)  { oih[i] = (__bf16)ih[i]; ohh[i] = (__bf16)hh[i]; }
  if (i < 128 * 160) { int o = i / 160, k = i % 160; oe1[i] = (k < 132) ? (__bf16)e1[o * 132 + k] : (__bf16)0.0f; }
  if (i < 64 * 96)   { int o = i / 96,  k = i % 96;  op1[i] = (k < 65)  ? (__bf16)p1[o * 65 + k]  : (__bf16)0.0f; }
}

__global__ void k_f2bf(const float* __restrict__ src, __bf16* __restrict__ dst, long n) {
  long i = (long)blockIdx.x * blockDim.x + threadIdx.x;
  if (i < n) dst[i] = (__bf16)src[i];
}

// ---------------- GCN linear (16 -> 64, K padded to 32), bf16 output ----------------
__global__ void k_gcn_linear(const float* __restrict__ xt, const __bf16* __restrict__ W,
                             const float* __restrict__ b, __bf16* __restrict__ h, int ntiles) {
  int wid = blockIdx.x * (blockDim.x >> 5) + (threadIdx.x >> 5);
  if (wid >= ntiles) return;
  int lane = threadIdx.x & 31, half = lane >> 4, r = lane & 15;
  int row0 = wid * 16;
  const float* xr = xt + (long)(row0 + r) * 16 + half * 8;   // run k = half*8 + {0..7}
  v16bf A;
#pragma unroll
  for (int q = 0; q < 8; ++q) { A[q] = (__bf16)xr[q]; A[8 + q] = (__bf16)0.0f; }  // k>=16 padded
#pragma unroll
  for (int j = 0; j < 4; ++j) {
    v16bf B = loadB(W, 32, j * 16 + r, 0, half);
    v8f acc = {};
    acc = wmma_bf16(A, B, acc);
    float bb = b[j * 16 + r];
#pragma unroll
    for (int v = 0; v < 8; ++v)
      h[(long)(row0 + half * 8 + v) * 64 + j * 16 + r] = (__bf16)(acc[v] + bb);
  }
}

// ---------------- CSR gather: gcn[n] = dinv[n]^2*h[n] + sum_e nrm[e]*h[src[e]] -------------
__global__ void k_gather(const __bf16* __restrict__ hbf, const float* __restrict__ dinv,
                         const int* __restrict__ offs, const int* __restrict__ srcs,
                         const float* __restrict__ nrm, __bf16* __restrict__ out, int N) {
  int idx = blockIdx.x * blockDim.x + threadIdx.x;
  if (idx >= N * 16) return;
  int n = idx >> 4, c = (idx & 15) * 4;
  float d = dinv[n], w0 = d * d;
  v4bf hv = *(const v4bf*)(hbf + (long)n * 64 + c);
  float a0 = w0 * (float)hv[0], a1 = w0 * (float)hv[1];
  float a2 = w0 * (float)hv[2], a3 = w0 * (float)hv[3];
  int b0 = offs[n], b1 = offs[n + 1];
  for (int k = b0; k < b1; ++k) {
    int s = srcs[k]; float w = nrm[k];
    v4bf x = *(const v4bf*)(hbf + (long)s * 64 + c);
    a0 += w * (float)x[0]; a1 += w * (float)x[1];
    a2 += w * (float)x[2]; a3 += w * (float)x[3];
  }
  v4bf o; o[0] = (__bf16)a0; o[1] = (__bf16)a1; o[2] = (__bf16)a2; o[3] = (__bf16)a3;
  *(v4bf*)(out + (long)n * 64 + c) = o;
}

// ---------------- fused WMMA GRU step (in-place h update, 16 nodes / wave) ----------------
__global__ void k_gru(const __bf16* __restrict__ x, float* __restrict__ hs,
                      const __bf16* __restrict__ Wih, const __bf16* __restrict__ Whh,
                      const float* __restrict__ bih, const float* __restrict__ bhh, int ntiles) {
  int wid = blockIdx.x * (blockDim.x >> 5) + (threadIdx.x >> 5);
  if (wid >= ntiles) return;
  int lane = threadIdx.x & 31, half = lane >> 4, r = lane & 15;
  int row0 = wid * 16;
  v16bf Ax[2], Ah[2];
#pragma unroll
  for (int ks = 0; ks < 2; ++ks) {
    const __bf16* px = x + (long)(row0 + r) * 64 + ks * 32 + half * 8;
    ((uint4*)&Ax[ks])[0] = *(const uint4*)px;
    ((uint4*)&Ax[ks])[1] = *(const uint4*)(px + 16);
    const float* ph = hs + (long)(row0 + r) * 64 + ks * 32 + half * 8;
#pragma unroll
    for (int q = 0; q < 8; ++q) { Ah[ks][q] = (__bf16)ph[q]; Ah[ks][8 + q] = (__bf16)ph[16 + q]; }
  }
#pragma unroll
  for (int g = 0; g < 4; ++g) {
    int cR = g * 16 + r, cZ = 64 + g * 16 + r, cN = 128 + g * 16 + r;
    v8f air = {}, ahr = {}, aiz = {}, ahz = {}, ain = {}, ahn = {};
#pragma unroll
    for (int ks = 0; ks < 2; ++ks) {
      air = wmma_bf16(Ax[ks], loadB(Wih, 64, cR, ks, half), air);
      ahr = wmma_bf16(Ah[ks], loadB(Whh, 64, cR, ks, half), ahr);
      aiz = wmma_bf16(Ax[ks], loadB(Wih, 64, cZ, ks, half), aiz);
      ahz = wmma_bf16(Ah[ks], loadB(Whh, 64, cZ, ks, half), ahz);
      ain = wmma_bf16(Ax[ks], loadB(Wih, 64, cN, ks, half), ain);
      ahn = wmma_bf16(Ah[ks], loadB(Whh, 64, cN, ks, half), ahn);
    }
    int ch = g * 16 + r;
#pragma unroll
    for (int v = 0; v < 8; ++v) {
      long node = row0 + half * 8 + v;
      float ir = air[v] + bih[ch],       hr = ahr[v] + bhh[ch];
      float iz = aiz[v] + bih[64 + ch],  hz = ahz[v] + bhh[64 + ch];
      float in = ain[v] + bih[128 + ch], hn = ahn[v] + bhh[128 + ch];
      float rg = 1.0f / (1.0f + expf(-(ir + hr)));
      float zg = 1.0f / (1.0f + expf(-(iz + hz)));
      float ng = tanhf(in + rg * hn);
      float hp = hs[node * 64 + ch];
      hs[node * 64 + ch] = (1.0f - zg) * ng + zg * hp;
    }
  }
}

// run-of-8 A-fragment loader for edge features [emb[s](64) | emb[d](64) | attr(4) | pad]
__device__ __forceinline__ void edgeRun8(v16bf& F, int o, int s0,
                                         const __bf16* __restrict__ embS,
                                         const __bf16* __restrict__ embD,
                                         const float* __restrict__ attr) {
  if (s0 + 8 <= 64) {
    ((uint4*)&F)[o] = *(const uint4*)(embS + s0);
  } else if (s0 + 8 <= 128) {
    ((uint4*)&F)[o] = *(const uint4*)(embD + (s0 - 64));
  } else if (s0 == 128) {
#pragma unroll
    for (int q = 0; q < 8; ++q) F[o * 8 + q] = (q < 4) ? (__bf16)attr[q] : (__bf16)0.0f;
  } else {
#pragma unroll
    for (int q = 0; q < 8; ++q) F[o * 8 + q] = (__bf16)0.0f;
  }
}

// ---------------- edge MLP: relu([emb[s],emb[d],attr] @ e1^T + b1) @ e2^T + b2 -------------
__global__ void k_edge(const __bf16* __restrict__ emb, const float* __restrict__ eattr,
                       const int* __restrict__ row, const int* __restrict__ col,
                       const __bf16* __restrict__ W1, const float* __restrict__ b1,
                       const float* __restrict__ W2, const float* __restrict__ b2,
                       float* __restrict__ pred, float* __restrict__ nflow, int ntiles) {
  int wid = blockIdx.x * (blockDim.x >> 5) + (threadIdx.x >> 5);
  if (wid >= ntiles) return;
  int lane = threadIdx.x & 31, half = lane >> 4, r = lane & 15;
  int e0 = wid * 16, er = e0 + r;
  int s = row[er], d = col[er];
  const __bf16* embS = emb + (long)s * 64;
  const __bf16* embD = emb + (long)d * 64;
  const float*  attr = eattr + (long)er * 4;
  v16bf A[5];
#pragma unroll
  for (int ks = 0; ks < 5; ++ks) {
    int r0 = ks * 32 + half * 8;
    edgeRun8(A[ks], 0, r0, embS, embD, attr);
    edgeRun8(A[ks], 1, r0 + 16, embS, embD, attr);
  }
  float part[8];
#pragma unroll
  for (int v = 0; v < 8; ++v) part[v] = 0.0f;
#pragma unroll
  for (int j = 0; j < 8; ++j) {
    v8f acc = {};
#pragma unroll
    for (int ks = 0; ks < 5; ++ks)
      acc = wmma_bf16(A[ks], loadB(W1, 160, j * 16 + r, ks, half), acc);
    int c = j * 16 + r;
    float w2 = W2[c], bb = b1[c];
#pragma unroll
    for (int v = 0; v < 8; ++v) part[v] += fmaxf(acc[v] + bb, 0.0f) * w2;
  }
#pragma unroll
  for (int m = 1; m < 16; m <<= 1)
#pragma unroll
    for (int v = 0; v < 8; ++v) part[v] += __shfl_xor(part[v], m, 16);
  if (r == 0) {
#pragma unroll
    for (int v = 0; v < 8; ++v) {
      int e = e0 + half * 8 + v;
      float p = part[v] + b2[0];
      pred[e] = p;
      atomAddF(&nflow[col[e]], p);
    }
  }
}

// ---------------- node MLP: relu([emb,flow] @ p1^T + b1) @ p2^T + b2 ----------------
__global__ void k_node(const __bf16* __restrict__ emb, const float* __restrict__ flow,
                       const __bf16* __restrict__ W1, const float* __restrict__ b1,
                       const float* __restrict__ W2, const float* __restrict__ b2,
                       float* __restrict__ pred, int ntiles) {
  int wid = blockIdx.x * (blockDim.x >> 5) + (threadIdx.x >> 5);
  if (wid >= ntiles) return;
  int lane = threadIdx.x & 31, half = lane >> 4, r = lane & 15;
  int n0 = wid * 16, nr = n0 + r;
  const __bf16* embN = emb + (long)nr * 64;
  float fl = flow[nr];
  v16bf A[3];
#pragma unroll
  for (int ks = 0; ks < 3; ++ks) {
#pragma unroll
    for (int o = 0; o < 2; ++o) {
      int s0 = ks * 32 + half * 8 + o * 16;
      if (s0 + 8 <= 64) {
        ((uint4*)&A[ks])[o] = *(const uint4*)(embN + s0);
      } else if (s0 == 64) {
#pragma unroll
        for (int q = 0; q < 8; ++q) A[ks][o * 8 + q] = (q == 0) ? (__bf16)fl : (__bf16)0.0f;
      } else {
#pragma unroll
        for (int q = 0; q < 8; ++q) A[ks][o * 8 + q] = (__bf16)0.0f;
      }
    }
  }
  float part[8];
#pragma unroll
  for (int v = 0; v < 8; ++v) part[v] = 0.0f;
#pragma unroll
  for (int j = 0; j < 4; ++j) {
    v8f acc = {};
#pragma unroll
    for (int ks = 0; ks < 3; ++ks)
      acc = wmma_bf16(A[ks], loadB(W1, 96, j * 16 + r, ks, half), acc);
    int c = j * 16 + r;
    float w2 = W2[c], bb = b1[c];
#pragma unroll
    for (int v = 0; v < 8; ++v) part[v] += fmaxf(acc[v] + bb, 0.0f) * w2;
  }
#pragma unroll
  for (int m = 1; m < 16; m <<= 1)
#pragma unroll
    for (int v = 0; v < 8; ++v) part[v] += __shfl_xor(part[v], m, 16);
  if (r == 0) {
#pragma unroll
    for (int v = 0; v < 8; ++v) pred[n0 + half * 8 + v] = part[v] + b2[0];
  }
}

// ---------------- host ----------------
extern "C" void kernel_launch(void* const* d_in, const int* in_sizes, int n_in,
                              void* d_out, int out_size, void* d_ws, size_t ws_size,
                              hipStream_t stream) {
  (void)n_in; (void)ws_size;
  const float* x_seq = (const float*)d_in[0];
  const float* eattr = (const float*)d_in[1];
  const float* gcn_W = (const float*)d_in[2];
  const float* gcn_b = (const float*)d_in[3];
  const float* W_ih  = (const float*)d_in[4];
  const float* W_hh  = (const float*)d_in[5];
  const float* b_ih  = (const float*)d_in[6];
  const float* b_hh  = (const float*)d_in[7];
  const float* e1_W  = (const float*)d_in[8];
  const float* e1_b  = (const float*)d_in[9];
  const float* e2_W  = (const float*)d_in[10];
  const float* e2_b  = (const float*)d_in[11];
  const float* p1_W  = (const float*)d_in[12];
  const float* p1_b  = (const float*)d_in[13];
  const float* p2_W  = (const float*)d_in[14];
  const float* p2_b  = (const float*)d_in[15];
  const int*   eidx  = (const int*)d_in[16];

  const int E = in_sizes[16] / 2;
  const int N = out_size - E;
  const int T = in_sizes[0] / (N * 16);
  const int* row = eidx;
  const int* col = eidx + E;

  float* out_node = (float*)d_out;      // [N]
  float* out_edge = (float*)d_out + N;  // [E]

  auto al = [](size_t x) { return (x + 255) & ~(size_t)255; };
  char* base = (char*)d_ws;
  size_t off = 0;
  float*  dinv  = (float*)(base + off);  off += al((size_t)N * 4);
  float*  hstat = (float*)(base + off);  off += al((size_t)N * 64 * 4);
  __bf16* hbf   = (__bf16*)(base + off); off += al((size_t)N * 64 * 2);
  __bf16* gcnbf = (__bf16*)(base + off); off += al((size_t)N * 64 * 2);
  __bf16* embbf = (__bf16*)(base + off); off += al((size_t)N * 64 * 2);
  float*  nflow = (float*)(base + off);  off += al((size_t)N * 4);
  int*    cnt   = (int*)(base + off);    off += al((size_t)N * 4);
  int*    loc   = (int*)(base + off);    off += al((size_t)N * 4);
  int*    cur   = (int*)(base + off);    off += al((size_t)N * 4);
  int*    offs  = (int*)(base + off);    off += al((size_t)(N + 1) * 4);
  int*    bsum  = (int*)(base + off);    off += al((size_t)SCAN_B * 4);
  int*    srcs  = (int*)(base + off);    off += al((size_t)E * 4);
  float*  nrm   = (float*)(base + off);  off += al((size_t)E * 4);
  __bf16* wg    = (__bf16*)(base + off); off += al((size_t)64 * 32 * 2);
  __bf16* wih   = (__bf16*)(base + off); off += al((size_t)192 * 64 * 2);
  __bf16* whh   = (__bf16*)(base + off); off += al((size_t)192 * 64 * 2);
  __bf16* we1   = (__bf16*)(base + off); off += al((size_t)128 * 160 * 2);
  __bf16* wp1   = (__bf16*)(base + off); off += al((size_t)64 * 96 * 2);

  const int NT = N / 16;
  const int ET = E / 16;
  const int WPB = 8;
  const int gN = (NT + WPB - 1) / WPB;
  const int gE = (ET + WPB - 1) / WPB;
  const long n64 = (long)N * 64;
  const int nb = (N + SCAN_B - 1) / SCAN_B;

  // weight prep (padded bf16 copies)
  k_prep<<<(128 * 160 + 255) / 256, 256, 0, stream>>>(gcn_W, W_ih, W_hh, e1_W, p1_W,
                                                      wg, wih, whh, we1, wp1);
  // degrees + dinv
  k_filli<<<(N + 255) / 256, 256, 0, stream>>>(cnt, 0, N);
  k_cnt<<<4096, 256, 0, stream>>>(col, cnt, E);
  k_dinv<<<(N + 255) / 256, 256, 0, stream>>>(cnt, dinv, N);
  // CSR offsets via exclusive scan
  k_scan1<<<nb, SCAN_B, 0, stream>>>(cnt, loc, bsum, N);
  k_scan2<<<1, SCAN_B, 0, stream>>>(bsum, nb);
  k_scan3<<<(N + 255) / 256, 256, 0, stream>>>(loc, bsum, cnt, offs, N);
  // CSR fill
  k_filli<<<(N + 255) / 256, 256, 0, stream>>>(cur, 0, N);
  k_csrfill<<<4096, 256, 0, stream>>>(row, col, dinv, offs, cur, srcs, nrm, E);
  // init state
  k_fill<<<(int)((n64 + 255) / 256), 256, 0, stream>>>(hstat, 0.0f, n64);
  k_fill<<<(N + 255) / 256, 256, 0, stream>>>(nflow, 0.0f, N);

  for (int t = 0; t < T; ++t) {
    const float* xt = x_seq + (size_t)t * N * 16;
    k_gcn_linear<<<gN, 256, 0, stream>>>(xt, wg, gcn_b, hbf, NT);
    k_gather<<<(N * 16 + 255) / 256, 256, 0, stream>>>(hbf, dinv, offs, srcs, nrm, gcnbf, N);
    k_gru<<<gN, 256, 0, stream>>>(gcnbf, hstat, wih, whh, b_ih, b_hh, NT);
  }

  k_f2bf<<<(int)((n64 + 255) / 256), 256, 0, stream>>>(hstat, embbf, n64);
  k_edge<<<gE, 256, 0, stream>>>(embbf, eattr, row, col, we1, e1_b, e2_W, e2_b,
                                 out_edge, nflow, ET);
  k_node<<<gN, 256, 0, stream>>>(embbf, nflow, wp1, p1_b, p2_W, p2_b, out_node, NT);
}
